// LSTMTagger_63307817943320
// MI455X (gfx1250) — matrix operand order repaired and checked by reading, not compiled, not run on hardware
//
#include <hip/hip_runtime.h>
#include <hip/hip_bf16.h>

// ---------------- types for WMMA ----------------
typedef __attribute__((ext_vector_type(16))) _Float16 v16h;
typedef __attribute__((ext_vector_type(8)))  _Float16 v8h;
typedef __attribute__((ext_vector_type(8)))  float    v8f;

union V16 { v16h v; v8h h[2]; };

// ---------------- problem dims ----------------
#define BB    64
#define SS    512
#define NROW  (BB * SS)   // 32768 (b,s) rows, flat order == all the reference .view()s
#define LP    18          // padded word length
#define CE_D  10
#define LL    32          // char conv channels
#define WE_D  200
#define DIN   232         // WE + L
#define KIN   256         // DIN padded to 32
#define HH    200
#define KH    224         // H padded to 32
#define NG    800         // 4*H gate width
#define TWOH  400
#define KL    416         // 2H padded to 32
#define TT    64          // tag count

__device__ __forceinline__ float sigf(float x) { return 1.0f / (1.0f + expf(-x)); }

// =====================================================================
// Kernel 1: build X[NROW][KIN] f16 = [word_emb(200) | char_cnn_maxpool(32) | 0 pad]
// One wave per row.
// =====================================================================
__global__ __launch_bounds__(256) void k_prep_x(
    const int* __restrict__ char_ids, const int* __restrict__ word_ids,
    const float* __restrict__ word_emb, const float* __restrict__ char_emb,
    const float* __restrict__ conv_w, const float* __restrict__ conv_b,
    _Float16* __restrict__ X)
{
    __shared__ float s_ce[128 * CE_D];   // 1280 floats
    __shared__ float s_cw[LL * 30];      // 960
    __shared__ float s_cb[LL];

    int tid = threadIdx.x;
    for (int i = tid; i < 128 * CE_D; i += 256) s_ce[i] = char_emb[i];
    for (int i = tid; i < LL * 30;   i += 256) s_cw[i] = conv_w[i];
    if (tid < LL) s_cb[tid] = conv_b[tid];
    __syncthreads();

    const int t    = blockIdx.x * 8 + (tid >> 5);   // row 0..32767
    const int lane = tid & 31;

    int ids[LP];
#pragma unroll
    for (int i = 0; i < LP; ++i) ids[i] = char_ids[t * LP + i];

    // char CNN + maxpool: lane l computes channel l
    float best = -1e30f;
    for (int w = 0; w < 16; ++w) {
        float acc = s_cb[lane];
#pragma unroll
        for (int k = 0; k < 3; ++k) {
            const float* e  = &s_ce[ids[w + k] * CE_D];
            const float* ww = &s_cw[lane * 30 + k * CE_D];
#pragma unroll
            for (int d = 0; d < CE_D; ++d) acc += e[d] * ww[d];
        }
        best = fmaxf(best, acc);
    }
    _Float16* xr = X + (size_t)t * KIN;
    xr[WE_D + lane] = (_Float16)best;

    // word embedding gather
    const int wid = word_ids[t];
    const float* wr = word_emb + (size_t)wid * WE_D;
    for (int j = lane; j < WE_D; j += 32) xr[j] = (_Float16)wr[j];
    // zero pad 232..255
    if (lane < KIN - DIN) xr[DIN + lane] = (_Float16)0.0f;
}

// =====================================================================
// Kernel 2: f32 -> f16 with K padding (weights), generic
// =====================================================================
__global__ void k_cvt_pad(const float* __restrict__ src, _Float16* __restrict__ dst,
                          int rows, int ksrc, int kdst)
{
    int i = blockIdx.x * blockDim.x + threadIdx.x;
    if (i >= rows * kdst) return;
    int r = i / kdst, k = i - r * kdst;
    dst[i] = (k < ksrc) ? (_Float16)src[r * ksrc + k] : (_Float16)0.0f;
}

__global__ void k_bias(const float* __restrict__ a, const float* __restrict__ b,
                       float* __restrict__ dst, int n)
{
    int i = blockIdx.x * blockDim.x + threadIdx.x;
    if (i < n) dst[i] = a[i] + b[i];
}

// =====================================================================
// Kernel 3: persistent bidirectional LSTM. grid = {2 blocks: dir}, 800 threads.
// 25 waves x 8 tiles == 4x50 tiles of the [64, 800] gate matrix.
// gates in LDS f32, h in LDS f16, c in registers.
// Weight fragments are re-loaded from L2 each step. An asm-laundered integer
// offset (added to the original global pointers, preserving address-space
// inference -> global_load_b128, not flat) defeats LICM so nothing is spilled
// to scratch across the 512-step chain.
// =====================================================================
__global__ __launch_bounds__(800, 1) void k_lstm(
    const _Float16* __restrict__ X,          // [NROW, KIN]
    const _Float16* __restrict__ Wih_all,    // [2][NG, KIN]
    const _Float16* __restrict__ Whh_all,    // [2][NG, KH]
    const float*    __restrict__ bias_all,   // [2][NG]  (b_ih + b_hh)
    _Float16* __restrict__ OUT)              // [NROW, KL] f16, cols 0..399 = [hf|hb]
{
    __shared__ __align__(16) float    gates[BB * NG];     // 204800 B
    __shared__ __align__(16) _Float16 hbuf[BB * KH];      //  28672 B

    const int dir  = blockIdx.x;
    const size_t wihBase = (size_t)dir * NG * KIN;
    const size_t whhBase = (size_t)dir * NG * KH;
    const float* bias    = bias_all + (size_t)dir * NG;

    const int tid  = threadIdx.x;
    const int wv   = tid >> 5;
    const int lane = tid & 31;
    const int nl   = lane & 15;
    const int hi   = lane >> 4;

    // init: h = 0 (incl. K-pad cols, which stay zero forever), c = 0
    for (int i = tid; i < BB * KH; i += 800) hbuf[i] = (_Float16)0.0f;
    float c00, c01, c02, c03, c04, c05, c06, c07;
    float c08, c09, c10, c11, c12, c13, c14, c15;
    c00=c01=c02=c03=c04=c05=c06=c07=0.0f;
    c08=c09=c10=c11=c12=c13=c14=c15=0.0f;
    float* creg0[16] = {&c00,&c01,&c02,&c03,&c04,&c05,&c06,&c07,
                        &c08,&c09,&c10,&c11,&c12,&c13,&c14,&c15};
    // dir 0 zeroes the 400..415 pad columns of OUT once
    if (dir == 0) {
        for (int i = tid; i < NROW * (KL - TWOH); i += 800) {
            int r = i >> 4, c = i & 15;
            OUT[(size_t)r * KL + TWOH + c] = (_Float16)0.0f;
        }
    }
    __syncthreads();

    for (int step = 0; step < SS; ++step) {
        const int s = dir ? (SS - 1 - step) : step;

        // opaque zero offset: redefined by volatile asm every iteration, so
        // LICM cannot hoist weight fragment loads out of the 512-step loop
        // (prevents scratch spilling); pointer provenance is preserved so
        // the loads stay global_load (not flat_load).
        unsigned wzero = 0;
        asm volatile("" : "+v"(wzero));
        const _Float16* wih = Wih_all + wihBase + wzero;
        const _Float16* whh = Whh_all + whhBase + wzero;

        // ---- GEMM phase: gates = X_s @ Wih^T + h @ Whh^T + bias ----
#pragma unroll 1
        for (int i = 0; i < 8; ++i) {
            const int ti = wv * 8 + i;   // 0..199
            const int mt = ti & 3;       // batch tile 0..3
            const int nt = ti >> 2;      // gate tile 0..49

            v8f acc;
            const float bv = bias[nt * 16 + nl];
#pragma unroll
            for (int r = 0; r < 8; ++r) acc[r] = bv;

            const _Float16* ap = X   + (size_t)(s * BB + mt * 16 + nl) * KIN;
            const _Float16* bp = wih + (size_t)(nt * 16 + nl) * KIN;
#pragma unroll
            for (int ks = 0; ks < KIN / 32; ++ks) {
                V16 a, b;
                a.h[0] = *(const v8h*)(ap + ks * 32 + hi * 8);
                a.h[1] = *(const v8h*)(ap + ks * 32 + 16 + hi * 8);
                b.h[0] = *(const v8h*)(bp + ks * 32 + hi * 8);
                b.h[1] = *(const v8h*)(bp + ks * 32 + 16 + hi * 8);
                acc = __builtin_amdgcn_wmma_f32_16x16x32_f16(
                          false, a.v, false, b.v, (short)0, acc, false, false);
            }
            const _Float16* hp  = hbuf + (size_t)(mt * 16 + nl) * KH;
            const _Float16* bp2 = whh  + (size_t)(nt * 16 + nl) * KH;
#pragma unroll
            for (int ks = 0; ks < KH / 32; ++ks) {
                V16 a, b;
                a.h[0] = *(const v8h*)(hp  + ks * 32 + hi * 8);
                a.h[1] = *(const v8h*)(hp  + ks * 32 + 16 + hi * 8);
                b.h[0] = *(const v8h*)(bp2 + ks * 32 + hi * 8);
                b.h[1] = *(const v8h*)(bp2 + ks * 32 + 16 + hi * 8);
                acc = __builtin_amdgcn_wmma_f32_16x16x32_f16(
                          false, a.v, false, b.v, (short)0, acc, false, false);
            }
#pragma unroll
            for (int r = 0; r < 8; ++r)
                gates[(mt * 16 + r + hi * 8) * NG + nt * 16 + nl] = acc[r];
        }

        // prefetch next step's X slab (64 rows x 256 f16 = 32 KB) while we
        // wait at the barrier / run the cell phase.
        {
            const int s2 = dir ? (s - 1) : (s + 1);
            if (s2 >= 0 && s2 < SS && tid < 512) {
                const char* px = (const char*)(X + (size_t)s2 * BB * KIN) + tid * 64;
                __builtin_prefetch(px, 0, 0);
            }
        }
        __syncthreads();

        // ---- cell phase: i,f,g,o -> c,h ----  (c state lives in VGPRs)
#pragma unroll
        for (int k = 0; k < 16; ++k) {
            const int idx = tid + k * 800;       // 0..12799
            const int b = idx / HH, j = idx - b * HH;
            const float ig = gates[b * NG + j];
            const float fg = gates[b * NG + HH + j];
            const float gg = gates[b * NG + 2 * HH + j];
            const float og = gates[b * NG + 3 * HH + j];
            const float cn = sigf(fg) * (*creg0[k]) + sigf(ig) * tanhf(gg);
            const float h  = sigf(og) * tanhf(cn);
            *creg0[k] = cn;
            hbuf[b * KH + j] = (_Float16)h;
            OUT[(size_t)(s * BB + b) * KL + dir * HH + j] = (_Float16)h;
        }
        __syncthreads();
    }
}

// =====================================================================
// Kernel 4: TAG[NROW, 64] = LSTM[NROW, 400] @ out_w^T + out_b   (WMMA)
// one wave per 16x16 tile; 2048 m-tiles x 4 n-tiles = 8192 waves
// =====================================================================
__global__ __launch_bounds__(256) void k_tag(
    const _Float16* __restrict__ L,     // [NROW, KL]
    const _Float16* __restrict__ OW,    // [TT, KL]
    const float*    __restrict__ ob,    // [TT]
    float* __restrict__ TAG)            // [NROW, TT]
{
    const int tid  = threadIdx.x;
    const int gw   = blockIdx.x * 8 + (tid >> 5);  // 0..8191
    const int lane = tid & 31;
    const int nl   = lane & 15, hi = lane >> 4;
    const int mt   = gw >> 2, nt = gw & 3;

    v8f acc;
    const float bv = ob[nt * 16 + nl];
#pragma unroll
    for (int r = 0; r < 8; ++r) acc[r] = bv;

    const _Float16* ap = L  + (size_t)(mt * 16 + nl) * KL;
    const _Float16* bp = OW + (size_t)(nt * 16 + nl) * KL;
#pragma unroll 1
    for (int ks = 0; ks < KL / 32; ++ks) {
        V16 a, b;
        a.h[0] = *(const v8h*)(ap + ks * 32 + hi * 8);
        a.h[1] = *(const v8h*)(ap + ks * 32 + 16 + hi * 8);
        b.h[0] = *(const v8h*)(bp + ks * 32 + hi * 8);
        b.h[1] = *(const v8h*)(bp + ks * 32 + 16 + hi * 8);
        acc = __builtin_amdgcn_wmma_f32_16x16x32_f16(
                  false, a.v, false, b.v, (short)0, acc, false, false);
    }
#pragma unroll
    for (int r = 0; r < 8; ++r)
        TAG[(size_t)(mt * 16 + r + hi * 8) * TT + nt * 16 + nl] = acc[r];
}

// =====================================================================
// Kernel 5: log_softmax over the S axis (axis=1 of [B,S,T]).
// One thread per (b, t) column: 512 strided values.
// =====================================================================
__global__ void k_softmax(const float* __restrict__ TAG, float* __restrict__ out)
{
    const int g = blockIdx.x * blockDim.x + threadIdx.x;   // 0..4095
    const int b = g >> 6, t = g & 63;
    const size_t base = ((size_t)b * SS) * TT + t;

    float m = -1e30f;
    for (int s = 0; s < SS; ++s) m = fmaxf(m, TAG[base + (size_t)s * TT]);
    float sum = 0.0f;
    for (int s = 0; s < SS; ++s) sum += expf(TAG[base + (size_t)s * TT] - m);
    const float lse = m + logf(sum);
    for (int s = 0; s < SS; ++s) {
        const size_t i = base + (size_t)s * TT;
        out[i] = TAG[i] - lse;
    }
}

// =====================================================================
// host launcher
// =====================================================================
extern "C" void kernel_launch(void* const* d_in, const int* in_sizes, int n_in,
                              void* d_out, int out_size, void* d_ws, size_t ws_size,
                              hipStream_t stream) {
    (void)in_sizes; (void)n_in; (void)out_size; (void)ws_size;

    const int*   char_ids = (const int*)  d_in[0];
    const int*   word_ids = (const int*)  d_in[1];
    const float* word_emb = (const float*)d_in[2];
    const float* char_emb = (const float*)d_in[3];
    const float* conv_w   = (const float*)d_in[4];
    const float* conv_b   = (const float*)d_in[5];
    const float* w_ih_f   = (const float*)d_in[6];
    const float* w_hh_f   = (const float*)d_in[7];
    const float* b_ih_f   = (const float*)d_in[8];
    const float* b_hh_f   = (const float*)d_in[9];
    const float* w_ih_b   = (const float*)d_in[10];
    const float* w_hh_b   = (const float*)d_in[11];
    const float* b_ih_b   = (const float*)d_in[12];
    const float* b_hh_b   = (const float*)d_in[13];
    const float* out_w    = (const float*)d_in[14];
    const float* out_b    = (const float*)d_in[15];
    float* out = (float*)d_out;

    // workspace carve-up (bytes)
    char* ws = (char*)d_ws;
    size_t off = 0;
    _Float16* X     = (_Float16*)(ws + off); off += (size_t)NROW * KIN * 2;   // 16.8 MB
    _Float16* Wih   = (_Float16*)(ws + off); off += (size_t)2 * NG * KIN * 2; // 0.82 MB
    _Float16* Whh   = (_Float16*)(ws + off); off += (size_t)2 * NG * KH * 2;  // 0.72 MB
    float*    biasW = (float*)   (ws + off); off += (size_t)2 * NG * 4;
    _Float16* OW    = (_Float16*)(ws + off); off += (size_t)TT * KL * 2;
    _Float16* LOUT  = (_Float16*)(ws + off); off += (size_t)NROW * KL * 2;    // 27.3 MB
    float*    TAG   = (float*)   (ws + off); off += (size_t)NROW * TT * 4;    // 8.4 MB

    // 1) build X (char CNN + embeddings)
    k_prep_x<<<NROW / 8, 256, 0, stream>>>(char_ids, word_ids, word_emb, char_emb,
                                           conv_w, conv_b, X);
    // 2) weight conversion / padding
    auto grid = [](int n) { return (n + 255) / 256; };
    k_cvt_pad<<<grid(NG * KIN), 256, 0, stream>>>(w_ih_f, Wih,            NG, DIN,  KIN);
    k_cvt_pad<<<grid(NG * KIN), 256, 0, stream>>>(w_ih_b, Wih + NG * KIN, NG, DIN,  KIN);
    k_cvt_pad<<<grid(NG * KH ), 256, 0, stream>>>(w_hh_f, Whh,            NG, HH,   KH);
    k_cvt_pad<<<grid(NG * KH ), 256, 0, stream>>>(w_hh_b, Whh + NG * KH,  NG, HH,   KH);
    k_cvt_pad<<<grid(TT * KL ), 256, 0, stream>>>(out_w,  OW,             TT, TWOH, KL);
    k_bias<<<grid(NG), 256, 0, stream>>>(b_ih_f, b_hh_f, biasW,      NG);
    k_bias<<<grid(NG), 256, 0, stream>>>(b_ih_b, b_hh_b, biasW + NG, NG);

    // 3) bidirectional LSTM (persistent; block 0 = fwd, block 1 = bwd)
    k_lstm<<<2, 800, 0, stream>>>(X, Wih, Whh, biasW, LOUT);

    // 4) tag projection (WMMA)
    k_tag<<<(NROW / 16) * 4 / 8, 256, 0, stream>>>(LOUT, OW, out_b, TAG);

    // 5) log-softmax over seq axis
    k_softmax<<<16, 256, 0, stream>>>(TAG, out);
}
